// BERTBase_28698971472286
// MI455X (gfx1250) — matrix operand order repaired and checked by reading, not compiled, not run on hardware
//
#include <hip/hip_runtime.h>
#include <hip/hip_bf16.h>

typedef __attribute__((ext_vector_type(16))) __bf16 bf16x16;
typedef __attribute__((ext_vector_type(8)))  __bf16 bf16x8;
typedef __attribute__((ext_vector_type(8)))  float  f32x8;

union AFrag { bf16x16 v; bf16x8 h[2]; };
union U4BF  { uint4 u; __bf16 e[8]; };

#define B_DIM 32
#define S_DIM 512
#define E_DIM 768
#define H_DIM 12
#define DK_DIM 64
#define FF_DIM 3072
#define M_DIM (B_DIM * S_DIM)   // 16384

__device__ __forceinline__ __bf16 f2bf(float f) {
    unsigned u = __builtin_bit_cast(unsigned, f);
    u = (u + 0x7FFFu + ((u >> 16) & 1u)) >> 16;
    unsigned short s = (unsigned short)u;
    return __builtin_bit_cast(__bf16, s);
}

__device__ __forceinline__ float gelu_exact(float x) {
    return 0.5f * x * (1.0f + erff(x * 0.70710678118654752f));
}

// ---------------------------------------------------------------------------
// Weight transpose + fp32 -> bf16 convert:  W[K,N] (row-major) -> Wt[N,K] bf16
// block (32,8), grid (N/32, K/32)
// ---------------------------------------------------------------------------
__global__ void __launch_bounds__(256)
transpose_cvt(const float* __restrict__ W, __bf16* __restrict__ Wt,
              int K, int N) {
    __shared__ float t[32][33];
    int c = blockIdx.x * 32 + threadIdx.x;          // N index
    int rBase = blockIdx.y * 32;                    // K base
#pragma unroll
    for (int i = 0; i < 4; ++i) {
        int r = rBase + threadIdx.y + i * 8;
        t[threadIdx.y + i * 8][threadIdx.x] = W[(size_t)r * N + c];
    }
    __syncthreads();
#pragma unroll
    for (int i = 0; i < 4; ++i) {
        int n = blockIdx.x * 32 + threadIdx.y + i * 8;
        int k = rBase + threadIdx.x;
        Wt[(size_t)n * K + k] = f2bf(t[threadIdx.x][threadIdx.y + i * 8]);
    }
}

// ---------------------------------------------------------------------------
// fp32 -> bf16 elementwise convert
// ---------------------------------------------------------------------------
__global__ void __launch_bounds__(256)
cvt_bf16(const float* __restrict__ in, __bf16* __restrict__ out) {
    size_t i = (size_t)blockIdx.x * 256 + threadIdx.x;
    out[i] = f2bf(in[i]);
}

// ---------------------------------------------------------------------------
// Tiled bf16 WMMA GEMM:  C[M,N] = A[M,K] @ Bt[N,K]^T + bias
// block = 256 threads (8 waves), tile 128(M) x 64(N), K-step 64.
// Each wave computes a 32x32 sub-tile (2x2 WMMA tiles), 8 WMMAs per stage.
// EPI: 0 = fp32 out, 1 = GELU + bf16 out, 2 = +resid fp32 then bf16 out,
//      3 = bf16 out
// ---------------------------------------------------------------------------
template <int EPI>
__global__ void __launch_bounds__(256)
gemm_bf16(const __bf16* __restrict__ A,
          const __bf16* __restrict__ Bt,
          const float* __restrict__ bias,
          float* __restrict__ Cf,
          __bf16* __restrict__ Cb,
          const float* __restrict__ resid,
          int N, int K) {
    __shared__ __bf16 As[128 * 64];   // 16 KB
    __shared__ __bf16 Bs[64 * 64];    //  8 KB

    const int tid  = threadIdx.x;
    const int lane = tid & 31;
    const int wave = tid >> 5;
    const int m15  = lane & 15;
    const int half = lane >> 4;

    const int mBase = blockIdx.x * 128;
    const int nBase = blockIdx.y * 64;
    const int wm = (wave & 3) * 32;
    const int wn = (wave >> 2) * 32;

    f32x8 acc[2][2] = {};

    for (int k0 = 0; k0 < K; k0 += 64) {
        __syncthreads();
        // Stage A tile: 128 rows x 64 k  (1024 x 16B chunks, 4 per thread)
#pragma unroll
        for (int i = 0; i < 4; ++i) {
            int c = tid + i * 256;
            int r = c >> 3;
            int o = (c & 7) * 8;
            *(uint4*)&As[r * 64 + o] =
                *(const uint4*)&A[(size_t)(mBase + r) * K + k0 + o];
        }
        // Stage B tile: 64 rows (N) x 64 k  (512 x 16B chunks, 2 per thread)
#pragma unroll
        for (int i = 0; i < 2; ++i) {
            int c = tid + i * 256;
            int r = c >> 3;
            int o = (c & 7) * 8;
            *(uint4*)&Bs[r * 64 + o] =
                *(const uint4*)&Bt[(size_t)(nBase + r) * K + k0 + o];
        }
        __syncthreads();

#pragma unroll
        for (int kk = 0; kk < 2; ++kk) {
            AFrag a[2];
            bf16x16 b[2];
#pragma unroll
            for (int t = 0; t < 2; ++t) {
                int row = wm + t * 16 + m15;
                a[t].h[0] = *(const bf16x8*)&As[row * 64 + kk * 32 + half * 16];
                a[t].h[1] = *(const bf16x8*)&As[row * 64 + kk * 32 + half * 16 + 8];
            }
#pragma unroll
            for (int t = 0; t < 2; ++t) {
                int col = wn + t * 16 + m15;
                b[t] = *(const bf16x16*)&Bs[col * 64 + kk * 32 + half * 16];
            }
#pragma unroll
            for (int i = 0; i < 2; ++i)
#pragma unroll
                for (int j = 0; j < 2; ++j)
                    acc[i][j] = __builtin_amdgcn_wmma_f32_16x16x32_bf16(
                        false, a[i].v, false, b[j], (short)0, acc[i][j], false, false);
        }
    }

    // Epilogue
#pragma unroll
    for (int i = 0; i < 2; ++i) {
#pragma unroll
        for (int j = 0; j < 2; ++j) {
            int col = nBase + wn + j * 16 + m15;
            float bv = bias[col];
#pragma unroll
            for (int g = 0; g < 8; ++g) {
                int row = mBase + wm + i * 16 + g + 8 * half;
                size_t idx = (size_t)row * N + col;
                float v = acc[i][j][g] + bv;
                if (EPI == 0) {
                    Cf[idx] = v;
                } else if (EPI == 1) {
                    Cb[idx] = f2bf(gelu_exact(v));
                } else if (EPI == 2) {
                    Cb[idx] = f2bf(v + resid[idx]);
                } else {
                    Cb[idx] = f2bf(v);
                }
            }
        }
    }
}

// ---------------------------------------------------------------------------
// Flash-style WMMA attention.
// grid = B*H*(S/64) blocks, 128 threads (4 waves). Wave owns 16 query rows.
// Streams 32-key tiles; scores & P.V via bf16 WMMA; online softmax.
// ---------------------------------------------------------------------------
__global__ void __launch_bounds__(128)
attention_k(const __bf16* __restrict__ Qb,
            const __bf16* __restrict__ Kb,
            const __bf16* __restrict__ Vb,
            const int* __restrict__ mask,
            __bf16* __restrict__ attb) {
    __shared__ __bf16 Ks[32 * 64];   // [key][dk]
    __shared__ __bf16 Vt[64 * 32];   // [dk][key]
    __shared__ __bf16 Ps[4][16 * 32];

    const int tid  = threadIdx.x;
    const int lane = tid & 31;
    const int wave = tid >> 5;
    const int m15  = lane & 15;
    const int half = lane >> 4;

    const int bh   = blockIdx.x;
    const int b    = bh / (H_DIM * 8);
    const int rem  = bh % (H_DIM * 8);
    const int h    = rem / 8;
    const int qblk = rem % 8;

    const size_t rowBase = (size_t)b * S_DIM;
    const int col0 = h * DK_DIM;
    const int qRow0 = qblk * 64 + wave * 16;

    // Q fragments (16 rows x 64 dk) -- loaded once, reused all iterations
    AFrag qf[2];
#pragma unroll
    for (int c = 0; c < 2; ++c) {
        size_t row = rowBase + qRow0 + m15;
        qf[c].h[0] = *(const bf16x8*)&Qb[row * E_DIM + col0 + c * 32 + half * 16];
        qf[c].h[1] = *(const bf16x8*)&Qb[row * E_DIM + col0 + c * 32 + half * 16 + 8];
    }

    float m_loc[8], l_loc[8];
#pragma unroll
    for (int g = 0; g < 8; ++g) { m_loc[g] = -3.0e38f; l_loc[g] = 0.0f; }
    f32x8 oacc[4] = {};

    for (int kt = 0; kt < 16; ++kt) {
        const int keyBase = kt * 32;
        __syncthreads();
        // Stage K tile: 32 keys x 64 dk (256 x 16B chunks, 2 per thread)
#pragma unroll
        for (int i = 0; i < 2; ++i) {
            int c = tid + i * 128;
            int key = c >> 3;
            int o = (c & 7) * 8;
            *(uint4*)&Ks[key * 64 + o] =
                *(const uint4*)&Kb[(rowBase + keyBase + key) * E_DIM + col0 + o];
        }
        // Stage V tile transposed: Vt[dk][key], vectorized global reads
#pragma unroll
        for (int i = 0; i < 2; ++i) {
            int c = tid + i * 128;
            int key = c >> 3;
            int o = (c & 7) * 8;
            U4BF val;
            val.u = *(const uint4*)&Vb[(rowBase + keyBase + key) * E_DIM + col0 + o];
#pragma unroll
            for (int j = 0; j < 8; ++j)
                Vt[(o + j) * 32 + key] = val.e[j];
        }
        __syncthreads();

        // Scores: 16q x 32keys = 2 tiles, each K-reduced over dk in 2 chunks
        f32x8 sc[2] = {};
#pragma unroll
        for (int nt = 0; nt < 2; ++nt) {
#pragma unroll
            for (int c = 0; c < 2; ++c) {
                bf16x16 kb = *(const bf16x16*)&Ks[(nt * 16 + m15) * 64 + c * 32 + half * 16];
                sc[nt] = __builtin_amdgcn_wmma_f32_16x16x32_bf16(
                    false, qf[c].v, false, kb, (short)0, sc[nt], false, false);
            }
        }
        // Scale + mask (mask indexed by key column)
        int mk[2];
#pragma unroll
        for (int nt = 0; nt < 2; ++nt)
            mk[nt] = mask[b * S_DIM + keyBase + nt * 16 + m15];
#pragma unroll
        for (int nt = 0; nt < 2; ++nt)
#pragma unroll
            for (int g = 0; g < 8; ++g) {
                float s = sc[nt][g] * 0.125f;
                sc[nt][g] = (mk[nt] == 0) ? -1.0e30f : s;
            }

        // Online softmax: rows g+8*half live across the 16 lanes of this half
        float sc_l[8];
#pragma unroll
        for (int g = 0; g < 8; ++g) {
            float v = fmaxf(sc[0][g], sc[1][g]);
            v = fmaxf(v, __shfl_xor(v, 1));
            v = fmaxf(v, __shfl_xor(v, 2));
            v = fmaxf(v, __shfl_xor(v, 4));
            v = fmaxf(v, __shfl_xor(v, 8));
            float nm = fmaxf(m_loc[g], v);
            sc_l[g] = __expf(m_loc[g] - nm);
            m_loc[g] = nm;
        }
#pragma unroll
        for (int g = 0; g < 8; ++g) {
            float p0 = __expf(sc[0][g] - m_loc[g]);
            float p1 = __expf(sc[1][g] - m_loc[g]);
            sc[0][g] = p0; sc[1][g] = p1;
            float v = p0 + p1;
            v += __shfl_xor(v, 1);
            v += __shfl_xor(v, 2);
            v += __shfl_xor(v, 4);
            v += __shfl_xor(v, 8);
            l_loc[g] = l_loc[g] * sc_l[g] + v;
        }
#pragma unroll
        for (int d = 0; d < 4; ++d)
#pragma unroll
            for (int g = 0; g < 8; ++g)
                oacc[d][g] *= sc_l[g];

        // Re-pack P (C-layout -> A-layout) through wave-private LDS
#pragma unroll
        for (int nt = 0; nt < 2; ++nt)
#pragma unroll
            for (int g = 0; g < 8; ++g)
                Ps[wave][(g + 8 * half) * 32 + nt * 16 + m15] = f2bf(sc[nt][g]);

        AFrag pf;
        pf.h[0] = *(const bf16x8*)&Ps[wave][m15 * 32 + half * 16];
        pf.h[1] = *(const bf16x8*)&Ps[wave][m15 * 32 + half * 16 + 8];
#pragma unroll
        for (int d = 0; d < 4; ++d) {
            bf16x16 vb = *(const bf16x16*)&Vt[(d * 16 + m15) * 32 + half * 16];
            oacc[d] = __builtin_amdgcn_wmma_f32_16x16x32_bf16(
                false, pf.v, false, vb, (short)0, oacc[d], false, false);
        }
    }

    // Finalize: divide by row sums, store bf16 into [B*S, E] at head offset
#pragma unroll
    for (int d = 0; d < 4; ++d)
#pragma unroll
        for (int g = 0; g < 8; ++g) {
            size_t row = rowBase + qRow0 + g + 8 * half;
            float v = oacc[d][g] / l_loc[g];
            attb[row * E_DIM + col0 + d * 16 + m15] = f2bf(v);
        }
}

// ---------------------------------------------------------------------------
// Row LayerNorm over E=768. block=256, grid=M. Optional residual add and
// optional bf16 secondary output.
// ---------------------------------------------------------------------------
__global__ void __launch_bounds__(256)
layernorm_k(const float* __restrict__ in,
            const float* __restrict__ resid,
            const float* __restrict__ gamma,
            const float* __restrict__ beta,
            float* __restrict__ outf,
            __bf16* __restrict__ outb) {
    __shared__ float red[256];
    const int tid = threadIdx.x;
    const size_t row = blockIdx.x;
    const size_t base = row * E_DIM;

    float v[3];
#pragma unroll
    for (int i = 0; i < 3; ++i) {
        int e = tid + i * 256;
        v[i] = in[base + e];
        if (resid) v[i] += resid[base + e];
    }
    float s = v[0] + v[1] + v[2];
    red[tid] = s;
    __syncthreads();
    for (int o = 128; o > 0; o >>= 1) {
        if (tid < o) red[tid] += red[tid + o];
        __syncthreads();
    }
    float mu = red[0] * (1.0f / 768.0f);
    __syncthreads();

    float q = 0.0f;
#pragma unroll
    for (int i = 0; i < 3; ++i) { float d = v[i] - mu; q += d * d; }
    red[tid] = q;
    __syncthreads();
    for (int o = 128; o > 0; o >>= 1) {
        if (tid < o) red[tid] += red[tid + o];
        __syncthreads();
    }
    float var = red[0] * (1.0f / 768.0f);
    float rstd = rsqrtf(var + 1e-12f);

#pragma unroll
    for (int i = 0; i < 3; ++i) {
        int e = tid + i * 256;
        float o = (v[i] - mu) * rstd * gamma[e] + beta[e];
        outf[base + e] = o;
        if (outb) outb[base + e] = f2bf(o);
    }
}

// ---------------------------------------------------------------------------
extern "C" void kernel_launch(void* const* d_in, const int* in_sizes, int n_in,
                              void* d_out, int out_size, void* d_ws, size_t ws_size,
                              hipStream_t stream) {
    const float* x    = (const float*)d_in[0];
    const int*   mask = (const int*)  d_in[1];
    const float* Wq   = (const float*)d_in[2];
    const float* bq   = (const float*)d_in[3];
    const float* Wk   = (const float*)d_in[4];
    const float* bk   = (const float*)d_in[5];
    const float* Wv   = (const float*)d_in[6];
    const float* bv   = (const float*)d_in[7];
    const float* Wo   = (const float*)d_in[8];
    const float* bo   = (const float*)d_in[9];
    const float* Wso  = (const float*)d_in[10];
    const float* bso  = (const float*)d_in[11];
    const float* gso  = (const float*)d_in[12];
    const float* beso = (const float*)d_in[13];
    const float* Wi   = (const float*)d_in[14];
    const float* bi   = (const float*)d_in[15];
    const float* Wout = (const float*)d_in[16];
    const float* bout = (const float*)d_in[17];
    const float* gout = (const float*)d_in[18];
    const float* beout= (const float*)d_in[19];

    char* ws = (char*)d_ws;
    size_t off = 0;
    auto alloc = [&](size_t bytes) -> void* {
        void* p = ws + off;
        off += (bytes + 255) & ~(size_t)255;
        return p;
    };

    const size_t M = M_DIM, E = E_DIM, FF = FF_DIM;

    __bf16* WqT   = (__bf16*)alloc(E * E * 2);
    __bf16* WkT   = (__bf16*)alloc(E * E * 2);
    __bf16* WvT   = (__bf16*)alloc(E * E * 2);
    __bf16* WoT   = (__bf16*)alloc(E * E * 2);
    __bf16* WsoT  = (__bf16*)alloc(E * E * 2);
    __bf16* WiT   = (__bf16*)alloc(E * FF * 2);   // [FF, E]
    __bf16* WoutT = (__bf16*)alloc(FF * E * 2);   // [E, FF]
    __bf16* xbf   = (__bf16*)alloc(M * E * 2);
    __bf16* Qbf   = (__bf16*)alloc(M * E * 2);
    __bf16* Kbf   = (__bf16*)alloc(M * E * 2);
    __bf16* Vbf   = (__bf16*)alloc(M * E * 2);
    float*  h     = (float*) alloc(M * E * 4);
    __bf16* hbf   = (__bf16*)alloc(M * E * 2);
    __bf16* ffbf  = (__bf16*)alloc(M * FF * 2);
    // Region reuse (stream-serialized lifetimes):
    __bf16* attbf  = xbf;            // free after QKV GEMMs
    __bf16* sxabf  = Qbf;            // free after attention
    float*  hpre   = (float*)Kbf;    // spans Kbf+Vbf (exactly M*E*4 bytes)
    float*  outpre = (float*)xbf;    // spans xbf+Qbf (exactly M*E*4 bytes)

    dim3 tb(32, 8);
    transpose_cvt<<<dim3(E / 32, E / 32),  tb, 0, stream>>>(Wq,   WqT,   E,  E);
    transpose_cvt<<<dim3(E / 32, E / 32),  tb, 0, stream>>>(Wk,   WkT,   E,  E);
    transpose_cvt<<<dim3(E / 32, E / 32),  tb, 0, stream>>>(Wv,   WvT,   E,  E);
    transpose_cvt<<<dim3(E / 32, E / 32),  tb, 0, stream>>>(Wo,   WoT,   E,  E);
    transpose_cvt<<<dim3(E / 32, E / 32),  tb, 0, stream>>>(Wso,  WsoT,  E,  E);
    transpose_cvt<<<dim3(FF / 32, E / 32), tb, 0, stream>>>(Wi,   WiT,   E,  FF);
    transpose_cvt<<<dim3(E / 32, FF / 32), tb, 0, stream>>>(Wout, WoutT, FF, E);

    cvt_bf16<<<(M * E) / 256, 256, 0, stream>>>(x, xbf);

    dim3 gE(M / 128, E / 64);    // (128, 12)
    dim3 gFF(M / 128, FF / 64);  // (128, 48)

    // QKV projections
    gemm_bf16<3><<<gE, 256, 0, stream>>>(xbf, WqT, bq, nullptr, Qbf, nullptr, E, E);
    gemm_bf16<3><<<gE, 256, 0, stream>>>(xbf, WkT, bk, nullptr, Kbf, nullptr, E, E);
    gemm_bf16<3><<<gE, 256, 0, stream>>>(xbf, WvT, bv, nullptr, Vbf, nullptr, E, E);

    // Attention (flash-style)
    attention_k<<<B_DIM * H_DIM * (S_DIM / 64), 128, 0, stream>>>(Qbf, Kbf, Vbf, mask, attbf);

    // att @ Wo + bo + x  -> bf16
    gemm_bf16<2><<<gE, 256, 0, stream>>>(attbf, WoT, bo, nullptr, sxabf, x, E, E);
    // (x+att) @ Wso + bso -> fp32, then LN -> h (fp32 + bf16)
    gemm_bf16<0><<<gE, 256, 0, stream>>>(sxabf, WsoT, bso, hpre, nullptr, nullptr, E, E);
    layernorm_k<<<M, 256, 0, stream>>>(hpre, nullptr, gso, beso, h, hbf);
    // GELU(h @ Wi + bi) -> bf16
    gemm_bf16<1><<<gFF, 256, 0, stream>>>(hbf, WiT, bi, nullptr, ffbf, nullptr, FF, E);
    // ff @ Wout + bout -> fp32, then LN(+h residual) -> d_out
    gemm_bf16<0><<<gE, 256, 0, stream>>>(ffbf, WoutT, bout, outpre, nullptr, nullptr, E, FF);
    layernorm_k<<<M, 256, 0, stream>>>(outpre, h, gout, beout, (float*)d_out, nullptr);
}
